// multilingual_speech_model_61375082659840
// MI455X (gfx1250) — matrix operand order repaired and checked by reading, not compiled, not run on hardware
//
#include <hip/hip_runtime.h>
#include <cstdint>
#include <cstddef>

// ---------------------------------------------------------------------------
// Types / WMMA helpers (CDNA5 gfx1250, wave32)
// ---------------------------------------------------------------------------
typedef _Float16 h16;
typedef __attribute__((ext_vector_type(16))) _Float16 v16h;
typedef __attribute__((ext_vector_type(8)))  _Float16 v8h;
typedef __attribute__((ext_vector_type(8)))  float    v8f;

#define DEV __device__ __forceinline__

DEV v8f wmma_f16(v16h a, v16h b, v8f c) {
  // D = A(16x32) * B(32x16) + C, f32 accumulate
  return __builtin_amdgcn_wmma_f32_16x16x32_f16(false, a, false, b, (short)0, c,
                                                false, false);
}

// A fragment: 16x32 f16 tile, row-major source with leading dim lda (halves).
// lane L holds row (L&15); K chunks: lanes<16 -> k 0..15, lanes>=16 -> k 16..31.
DEV v16h load_frag_a(const h16* p0, int lda) {
  int lane = threadIdx.x & 31;
  const h16* p = p0 + (size_t)(lane & 15) * lda + ((lane >> 4) << 4);
  union { v16h v; v8h h[2]; } u;
  u.h[0] = *(const v8h*)(p);
  u.h[1] = *(const v8h*)(p + 8);
  return u.v;
}

// B fragment: 32x16 f16 tile taken from weight matrix W[out,in] row-major
// (so B = W^T). lane L holds output-col (L&15), 16 consecutive k values.
DEV v16h load_frag_b(const h16* p0, int ldw) {
  int lane = threadIdx.x & 31;
  return *(const v16h*)(p0 + (size_t)(lane & 15) * ldw + ((lane >> 4) << 4));
}

DEV float sigf(float x) { return 1.0f / (1.0f + __expf(-x)); }

// ---------------------------------------------------------------------------
// Generic batched GEMM:  C[M,Nout] = act( A[M,K](f16) @ W[Nout,K]^T + bias )
// Block = 128 threads (4 waves, 2x2), block tile 64(M) x 128(N),
// per-wave 32x64 (2 A frags x 4 B frags -> 8 WMMA per k-step).
// Depth-2 fragment pipeline; k-loop unrolling capped to bound registers.
// ---------------------------------------------------------------------------
template <typename OutT, bool TANH>
__global__ void gemm_ws(const h16* __restrict__ A, const h16* __restrict__ W,
                        const float* __restrict__ bias, OutT* __restrict__ C,
                        int K, int lda, int ldw, int Nout, int ldc) {
  const int wave = threadIdx.x >> 5;
  const int lane = threadIdx.x & 31;
  const int rowBase = blockIdx.y * 64 + (wave >> 1) * 32;
  const int colBase = blockIdx.x * 128 + (wave & 1) * 64;
  v8f acc[2][4] = {};
  // prologue: k = 0
  v16h a0c = load_frag_a(A + (size_t)rowBase * lda, lda);
  v16h a1c = load_frag_a(A + (size_t)(rowBase + 16) * lda, lda);
  v16h bc[4];
#pragma unroll
  for (int j = 0; j < 4; j++)
    bc[j] = load_frag_b(W + (size_t)(colBase + j * 16) * ldw, ldw);
#pragma unroll 1
  for (int k = 32; k < K; k += 32) {
    v16h a0n = load_frag_a(A + (size_t)rowBase * lda + k, lda);
    v16h a1n = load_frag_a(A + (size_t)(rowBase + 16) * lda + k, lda);
    v16h bn[4];
#pragma unroll
    for (int j = 0; j < 4; j++)
      bn[j] = load_frag_b(W + (size_t)(colBase + j * 16) * ldw + k, ldw);
#pragma unroll
    for (int j = 0; j < 4; j++) {
      acc[0][j] = wmma_f16(a0c, bc[j], acc[0][j]);
      acc[1][j] = wmma_f16(a1c, bc[j], acc[1][j]);
    }
    a0c = a0n; a1c = a1n;
#pragma unroll
    for (int j = 0; j < 4; j++) bc[j] = bn[j];
  }
#pragma unroll
  for (int j = 0; j < 4; j++) {
    acc[0][j] = wmma_f16(a0c, bc[j], acc[0][j]);
    acc[1][j] = wmma_f16(a1c, bc[j], acc[1][j]);
  }
  const int cw = lane & 15;
  const int rb = (lane >> 4) << 3;
#pragma unroll
  for (int mi = 0; mi < 2; mi++) {
#pragma unroll
    for (int j = 0; j < 4; j++) {
      int col = colBase + j * 16 + cw;
      if (col < Nout) {
        float bv = bias[col];
#pragma unroll
        for (int v = 0; v < 8; v++) {
          float val = acc[mi][j][v] + bv;
          if (TANH) val = tanhf(val);
          C[(size_t)(rowBase + mi * 16 + rb + v) * ldc + col] = (OutT)val;
        }
      }
    }
  }
}

// ---------------------------------------------------------------------------
// Persistent sequential LSTM layer (one direction per block; grid = 2).
// X = precomputed  x@Wih^T + b  gates  [S*32, 2048] f32.
// Per step: g = h@Whh^T (WMMA, h in LDS) + X[t]; cell update; h -> LDS + hcat.
// c lives in registers (16 elements / thread, 1024 threads = 32x512).
// Per k-step: one clause of fragment loads, then 8 WMMAs (no explicit double
// buffer -> no spills; hardware load counters overlap iterations).
// ---------------------------------------------------------------------------
__global__ void lstm_seq(const float* __restrict__ Xf, const float* __restrict__ Xb,
                         const h16* __restrict__ WhhF, const h16* __restrict__ WhhB,
                         h16* __restrict__ hcat, int S) {
  extern __shared__ __align__(16) char smem[];
  h16* h_sh = (h16*)smem;                   // 32 x 520
  h16* g_sh = (h16*)(smem + 32 * 520 * 2);  // 32 x 2064
  const int dir = blockIdx.x;
  const float* X = dir ? Xb : Xf;
  const h16* Whh = dir ? WhhB : WhhF;
  const int colOff = dir * 512;
  const int tid = threadIdx.x;
  const int wave = tid >> 5;
  const int lane = tid & 31;
  for (int i = tid; i < 32 * 520; i += 1024) h_sh[i] = (h16)0.0f;
  float creg[16];
#pragma unroll
  for (int k = 0; k < 16; k++) creg[k] = 0.0f;
  __syncthreads();
  const int mtile = wave & 1;       // 2 M tiles (32 rows)
  const int ngrp = wave >> 1;       // 16 groups x 128 cols
  const int cw = lane & 15;
  const int rb = (lane >> 4) << 3;
  const h16* Wbase = Whh + (size_t)(ngrp * 128) * 512;
#pragma unroll 1
  for (int t = 0; t < S; t++) {
    const int ts = dir ? (S - 1 - t) : t;
    if (t + 1 < S) {  // prefetch next step's precomputed gates
      const int tn = dir ? (S - 2 - t) : (t + 1);
      const float* pf = X + (size_t)tn * 32 * 2048;
      if (tid < 512) __builtin_prefetch(pf + (size_t)tid * 128, 0, 0);
    }
    // phase 1: g = h @ Whh^T   (32x512 @ 512x2048)
    v8f acc[8] = {};
#pragma unroll 1
    for (int kk = 0; kk < 16; kk++) {
      v16h a = load_frag_a(h_sh + mtile * 16 * 520 + kk * 32, 520);
      v16h b[8];
#pragma unroll
      for (int j = 0; j < 8; j++)
        b[j] = load_frag_b(Wbase + (size_t)(j * 16) * 512 + kk * 32, 512);
#pragma unroll
      for (int j = 0; j < 8; j++) acc[j] = wmma_f16(a, b[j], acc[j]);
    }
#pragma unroll
    for (int j = 0; j < 8; j++) {
      int col = ngrp * 128 + j * 16 + cw;
#pragma unroll
      for (int v = 0; v < 8; v++)
        g_sh[(mtile * 16 + rb + v) * 2064 + col] = (h16)acc[j][v];
    }
    __syncthreads();
    // phase 2: cell update
    const size_t xbase = (size_t)ts * 32 * 2048;
#pragma unroll
    for (int k = 0; k < 16; k++) {
      int e = tid + (k << 10);
      int row = e >> 9, col = e & 511;
      size_t rx = xbase + (size_t)row * 2048 + col;
      float gi = (float)g_sh[row * 2064 + col]        + X[rx];
      float gf = (float)g_sh[row * 2064 + col + 512]  + X[rx + 512];
      float gg = (float)g_sh[row * 2064 + col + 1024] + X[rx + 1024];
      float go = (float)g_sh[row * 2064 + col + 1536] + X[rx + 1536];
      float c = sigf(gf) * creg[k] + sigf(gi) * tanhf(gg);
      creg[k] = c;
      float h = sigf(go) * tanhf(c);
      h_sh[row * 520 + col] = (h16)h;
      hcat[((size_t)ts * 32 + row) * 1024 + colOff + col] = (h16)h;
    }
    __syncthreads();
  }
}

// ---------------------------------------------------------------------------
// Persistent bidirectional decoder (single block, both cells + ctx fused).
// Z* = precomputed  zt@Wih[:, :192]^T + b   [S*32,2048] f32.
// W*cat = [ Wih[:,192:256] | Whh ]  ->  [2048, 576] f16.
// LDS A buffers: cols 0..63 = ctx, 64..575 = h (per direction).
// ---------------------------------------------------------------------------
__global__ void lstm_decoder(const float* __restrict__ Zf, const float* __restrict__ Zb,
                             const h16* __restrict__ Wfcat, const h16* __restrict__ Wbcat,
                             const h16* __restrict__ ctxW, const float* __restrict__ ctx_b,
                             const float* __restrict__ h0, const float* __restrict__ c0,
                             h16* __restrict__ dec, int S) {
  extern __shared__ __align__(16) char smem[];
  h16* Af = (h16*)smem;            // 32 x 584
  h16* Ab = Af + 32 * 584;         // 32 x 584
  h16* g_sh = Ab + 32 * 584;       // 32 x 2064
  const int tid = threadIdx.x;
  const int wave = tid >> 5;
  const int lane = tid & 31;
  const int cw = lane & 15;
  const int rb = (lane >> 4) << 3;
  float cf[16], cb[16];
#pragma unroll
  for (int k = 0; k < 16; k++) {
    int e = tid + (k << 10);
    int row = e >> 9, col = e & 511;
    cf[k] = c0[col];
    cb[k] = c0[512 + col];
    Af[row * 584 + 64 + col] = (h16)h0[col];
    Ab[row * 584 + 64 + col] = (h16)h0[512 + col];
  }
  __syncthreads();
#pragma unroll 1
  for (int t = 0; t < S; t++) {
    if (t + 1 < S && tid < 512) {
      __builtin_prefetch(Zf + (size_t)(t + 1) * 32 * 2048 + (size_t)tid * 128, 0, 0);
      __builtin_prefetch(Zb + (size_t)(t + 1) * 32 * 2048 + (size_t)tid * 128, 0, 0);
    }
    // ctx = prev(=[hf|hb]) @ ctx_W^T + ctx_b   (32x1024 @ 1024x64), waves 0..7
    if (wave < 8) {
      const int mt = wave & 1, nt = wave >> 1;
      v8f acc = {};
#pragma unroll 1
      for (int kk = 0; kk < 32; kk++) {
        int k = kk * 32;
        const h16* ap = (k < 512) ? (Af + mt * 16 * 584 + 64 + k)
                                  : (Ab + mt * 16 * 584 + 64 + (k - 512));
        v16h a = load_frag_a(ap, 584);
        v16h b = load_frag_b(ctxW + (size_t)(nt * 16) * 1024 + k, 1024);
        acc = wmma_f16(a, b, acc);
      }
      int col = nt * 16 + cw;
      float bv = ctx_b[col];
#pragma unroll
      for (int v = 0; v < 8; v++) {
        h16 cv = (h16)(acc[v] + bv);
        int r = mt * 16 + rb + v;
        Af[r * 584 + col] = cv;
        Ab[r * 584 + col] = cv;
      }
    }
    __syncthreads();
#pragma unroll 1
    for (int d = 0; d < 2; d++) {
      const h16* Ar = d ? Ab : Af;
      h16* Aw = d ? Ab : Af;
      const h16* W = d ? Wbcat : Wfcat;
      const float* Z = d ? Zb : Zf;
      const int mt = wave & 1, ng = wave >> 1;
      const h16* Wbase = W + (size_t)(ng * 128) * 576;
      // gates GEMM, K = 576 = [ctx 64 | h 512]
      v8f acc[8] = {};
#pragma unroll 1
      for (int kk = 0; kk < 18; kk++) {
        v16h a = load_frag_a(Ar + mt * 16 * 584 + kk * 32, 584);
        v16h b[8];
#pragma unroll
        for (int j = 0; j < 8; j++)
          b[j] = load_frag_b(Wbase + (size_t)(j * 16) * 576 + kk * 32, 576);
#pragma unroll
        for (int j = 0; j < 8; j++) acc[j] = wmma_f16(a, b[j], acc[j]);
      }
#pragma unroll
      for (int j = 0; j < 8; j++) {
        int col = ng * 128 + j * 16 + cw;
#pragma unroll
        for (int v = 0; v < 8; v++)
          g_sh[(mt * 16 + rb + v) * 2064 + col] = (h16)acc[j][v];
      }
      __syncthreads();
      const size_t zbase = (size_t)t * 32 * 2048;
#pragma unroll
      for (int k = 0; k < 16; k++) {
        int e = tid + (k << 10);
        int row = e >> 9, col = e & 511;
        size_t rz = zbase + (size_t)row * 2048 + col;
        float gi = (float)g_sh[row * 2064 + col]        + Z[rz];
        float gf = (float)g_sh[row * 2064 + col + 512]  + Z[rz + 512];
        float gg = (float)g_sh[row * 2064 + col + 1024] + Z[rz + 1024];
        float go = (float)g_sh[row * 2064 + col + 1536] + Z[rz + 1536];
        float cp = d ? cb[k] : cf[k];
        float c = sigf(gf) * cp + sigf(gi) * tanhf(gg);
        if (d) cb[k] = c; else cf[k] = c;
        float h = sigf(go) * tanhf(c);
        Aw[row * 584 + 64 + col] = (h16)h;
        dec[((size_t)t * 32 + row) * 1024 + (size_t)d * 512 + col] = (h16)h;
      }
      __syncthreads();
    }
  }
}

// ---------------------------------------------------------------------------
// Small utility kernels
// ---------------------------------------------------------------------------
__global__ void fill_h16(h16* p, long n) {
  long i = (long)blockIdx.x * blockDim.x + threadIdx.x;
  if (i < n) p[i] = (h16)0.0f;
}
__global__ void fill_f32(float* p, long n, float v) {
  long i = (long)blockIdx.x * blockDim.x + threadIdx.x;
  if (i < n) p[i] = v;
}
__global__ void cvt_f32_to_f16(const float* __restrict__ src, h16* __restrict__ dst,
                               long rows, int cols, int src_ld, int src_off,
                               int dst_ld, int dst_off) {
  long i = (long)blockIdx.x * blockDim.x + threadIdx.x;
  long total = rows * (long)cols;
  if (i >= total) return;
  long r = i / cols;
  int c = (int)(i - r * cols);
  dst[r * dst_ld + dst_off + c] = (h16)src[r * src_ld + src_off + c];
}
// z = mu + exp(0.5*lv)*noise ; emit mu/log_var to d_out, zt (f16) for decoder
__global__ void latent_kernel(const float* __restrict__ mu_buf,
                              const float* __restrict__ lv_buf,
                              const float* __restrict__ noise,
                              float* __restrict__ mu_out, float* __restrict__ lv_out,
                              h16* __restrict__ zt) {
  long i = (long)blockIdx.x * blockDim.x + threadIdx.x;
  const long total = 3L * 800 * 32 * 64;
  if (i >= total) return;
  int e = (int)(i & 63);
  long r = i >> 6;
  int n = (int)(r & 31); r >>= 5;
  int s = (int)(r % 800);
  int k = (int)(r / 800);
  long src = ((long)(s * 32 + n)) * 192 + k * 64 + e;
  float m = mu_buf[src], lv = lv_buf[src];
  float z = m + __expf(0.5f * lv) * noise[i];
  mu_out[i] = m;
  lv_out[i] = lv;
  zt[src] = (h16)z;
}

// ---------------------------------------------------------------------------
// Host-side launch
// ---------------------------------------------------------------------------
static inline size_t al256(size_t x) { return (x + 255) & ~(size_t)255; }

extern "C" void kernel_launch(void* const* d_in, const int* in_sizes, int n_in,
                              void* d_out, int out_size, void* d_ws, size_t ws_size,
                              hipStream_t stream) {
  (void)in_sizes; (void)n_in; (void)out_size; (void)ws_size;
  constexpr int S = 800, NB = 32, M = S * NB;          // 25600 rows
  constexpr int IN = 321, INP = 352;                   // padded to 11*32
  constexpr int G = 2048, H = 512, KE = 192, DINC = 576;
  constexpr long Y_N = (long)S * NB * IN;              // 8,217,600
  constexpr long MU_N = 3L * S * NB * 64;              // 4,915,200

  const float* in_inp  = (const float*)d_in[0];
  const float* in_nz   = (const float*)d_in[1];
  // e0f 2-4, e0b 5-7, e1f 8-10, e1b 11-13, df 14-16, db 17-19
  // mu_W 20, mu_b 21, sig_W 22, sig_b 23, h0 24, c0 25, ctx 26-27,
  // feat 28-29, score 30-31

  char* ws = (char*)d_ws;
  size_t off = 0;
  auto alloc = [&](size_t bytes) -> char* { char* p = ws + off; off = al256(off + bytes); return p; };

  // ---- f16 arena (zero-filled each call; padded regions stay zero) ----
  size_t arena0 = off;
  h16* w_e0fWih = (h16*)alloc((size_t)G * INP * 2);
  h16* w_e0bWih = (h16*)alloc((size_t)G * INP * 2);
  h16* w_e0fWhh = (h16*)alloc((size_t)G * H * 2);
  h16* w_e0bWhh = (h16*)alloc((size_t)G * H * 2);
  h16* w_e1fWih = (h16*)alloc((size_t)G * 1024 * 2);
  h16* w_e1bWih = (h16*)alloc((size_t)G * 1024 * 2);
  h16* w_e1fWhh = (h16*)alloc((size_t)G * H * 2);
  h16* w_e1bWhh = (h16*)alloc((size_t)G * H * 2);
  h16* w_mu     = (h16*)alloc((size_t)256 * 1024 * 2);   // rows 192..255 zero pad
  h16* w_sig    = (h16*)alloc((size_t)256 * 1024 * 2);
  h16* w_ctx    = (h16*)alloc((size_t)64 * 1024 * 2);
  h16* w_dfz    = (h16*)alloc((size_t)G * KE * 2);
  h16* w_dbz    = (h16*)alloc((size_t)G * KE * 2);
  h16* w_dfcat  = (h16*)alloc((size_t)G * DINC * 2);
  h16* w_dbcat  = (h16*)alloc((size_t)G * DINC * 2);
  h16* w_feat   = (h16*)alloc((size_t)G * 1024 * 2);
  h16* w_score  = (h16*)alloc((size_t)384 * 2048 * 2);   // rows 321..383 zero pad
  h16* inp_h    = (h16*)alloc((size_t)M * INP * 2);      // cols 321..351 zero pad
  size_t arenaN = (off - arena0) / 2;                    // halves to zero-fill

  float* GATE_F = (float*)alloc((size_t)M * G * 4);
  float* GATE_B = (float*)alloc((size_t)M * G * 4);
  h16*   cat0   = (h16*)alloc((size_t)M * 1024 * 2);     // enc0 out, later dec out
  h16*   cat1   = (h16*)alloc((size_t)M * 1024 * 2);     // enc1 out
  float* mu_buf = (float*)alloc((size_t)M * KE * 4);
  float* lv_buf = (float*)alloc((size_t)M * KE * 4);
  h16*   zt_h   = (h16*)alloc((size_t)M * KE * 2);
  h16*   feat_h = (h16*)GATE_F;                          // reuse after decoder

  float* y_out  = (float*)d_out;
  float* mu_out = y_out + Y_N;
  float* lv_out = mu_out + MU_N;
  float* mk_out = lv_out + MU_N;

  const int TPB = 256;
  auto blocks = [&](long n) { return dim3((unsigned)((n + TPB - 1) / TPB)); };

  // 0) zero the f16 arena (covers all pads)
  fill_h16<<<blocks((long)arenaN), TPB, 0, stream>>>((h16*)(ws + arena0), (long)arenaN);

  // 1) weight / input conversions to f16
  auto cvt = [&](const void* src, h16* dst, long rows, int cols, int sld, int soff,
                 int dld, int doff) {
    cvt_f32_to_f16<<<blocks(rows * (long)cols), TPB, 0, stream>>>(
        (const float*)src, dst, rows, cols, sld, soff, dld, doff);
  };
  cvt(d_in[2],  w_e0fWih, G, IN, IN, 0, INP, 0);
  cvt(d_in[5],  w_e0bWih, G, IN, IN, 0, INP, 0);
  cvt(d_in[3],  w_e0fWhh, G, H, H, 0, H, 0);
  cvt(d_in[6],  w_e0bWhh, G, H, H, 0, H, 0);
  cvt(d_in[8],  w_e1fWih, G, 1024, 1024, 0, 1024, 0);
  cvt(d_in[11], w_e1bWih, G, 1024, 1024, 0, 1024, 0);
  cvt(d_in[9],  w_e1fWhh, G, H, H, 0, H, 0);
  cvt(d_in[12], w_e1bWhh, G, H, H, 0, H, 0);
  cvt(d_in[20], w_mu,  KE, 1024, 1024, 0, 1024, 0);
  cvt(d_in[22], w_sig, KE, 1024, 1024, 0, 1024, 0);
  cvt(d_in[26], w_ctx, 64, 1024, 1024, 0, 1024, 0);
  cvt(d_in[14], w_dfz, G, KE, 256, 0, KE, 0);
  cvt(d_in[17], w_dbz, G, KE, 256, 0, KE, 0);
  cvt(d_in[14], w_dfcat, G, 64, 256, 192, DINC, 0);   // ctx part of Wih
  cvt(d_in[15], w_dfcat, G, H, H, 0, DINC, 64);       // Whh
  cvt(d_in[17], w_dbcat, G, 64, 256, 192, DINC, 0);
  cvt(d_in[18], w_dbcat, G, H, H, 0, DINC, 64);
  cvt(d_in[28], w_feat, G, 1024, 1024, 0, 1024, 0);
  cvt(d_in[30], w_score, IN, 2048, 2048, 0, 2048, 0);
  cvt(in_inp, inp_h, M, IN, IN, 0, INP, 0);

  const dim3 gB(128);
  const dim3 gridG(16, M / 64);   // Nout 2048
  // 2) encoder layer 0: input projections + sequential scan
  gemm_ws<float, false><<<gridG, gB, 0, stream>>>(inp_h, w_e0fWih, (const float*)d_in[4],
                                                  GATE_F, INP, INP, INP, G, G);
  gemm_ws<float, false><<<gridG, gB, 0, stream>>>(inp_h, w_e0bWih, (const float*)d_in[7],
                                                  GATE_B, INP, INP, INP, G, G);
  lstm_seq<<<2, 1024, (32 * 520 + 32 * 2064) * 2, stream>>>(GATE_F, GATE_B, w_e0fWhh,
                                                            w_e0bWhh, cat0, S);
  // 3) encoder layer 1
  gemm_ws<float, false><<<gridG, gB, 0, stream>>>(cat0, w_e1fWih, (const float*)d_in[10],
                                                  GATE_F, 1024, 1024, 1024, G, G);
  gemm_ws<float, false><<<gridG, gB, 0, stream>>>(cat0, w_e1bWih, (const float*)d_in[13],
                                                  GATE_B, 1024, 1024, 1024, G, G);
  lstm_seq<<<2, 1024, (32 * 520 + 32 * 2064) * 2, stream>>>(GATE_F, GATE_B, w_e1fWhh,
                                                            w_e1bWhh, cat1, S);
  // 4) latent heads + reparameterization
  const dim3 gridL(2, M / 64);    // Nout 192 (padded 256)
  gemm_ws<float, false><<<gridL, gB, 0, stream>>>(cat1, w_mu, (const float*)d_in[21],
                                                  mu_buf, 1024, 1024, 1024, KE, KE);
  gemm_ws<float, false><<<gridL, gB, 0, stream>>>(cat1, w_sig, (const float*)d_in[23],
                                                  lv_buf, 1024, 1024, 1024, KE, KE);
  latent_kernel<<<blocks(MU_N), TPB, 0, stream>>>(mu_buf, lv_buf, in_nz, mu_out, lv_out,
                                                  zt_h);
  // 5) decoder: precompute z-part gates, then fused persistent decoder
  gemm_ws<float, false><<<gridG, gB, 0, stream>>>(zt_h, w_dfz, (const float*)d_in[16],
                                                  GATE_F, KE, KE, KE, G, G);
  gemm_ws<float, false><<<gridG, gB, 0, stream>>>(zt_h, w_dbz, (const float*)d_in[19],
                                                  GATE_B, KE, KE, KE, G, G);
  lstm_decoder<<<1, 1024, (2 * 32 * 584 + 32 * 2064) * 2, stream>>>(
      GATE_F, GATE_B, w_dfcat, w_dbcat, w_ctx, (const float*)d_in[27],
      (const float*)d_in[24], (const float*)d_in[25], cat0, S);
  // 6) output head: tanh(dec@feat^T+b) @ score^T + b
  gemm_ws<h16, true><<<gridG, gB, 0, stream>>>(cat0, w_feat, (const float*)d_in[29],
                                               feat_h, 1024, 1024, 1024, G, G);
  const dim3 gridS(3, M / 64);    // Nout 321 (padded 384)
  gemm_ws<float, false><<<gridS, gB, 0, stream>>>(feat_h, w_score, (const float*)d_in[31],
                                                  y_out, 2048, 2048, 2048, IN, IN);
  // 7) mask = ones
  fill_f32<<<blocks((long)S * NB), TPB, 0, stream>>>(mk_out, (long)S * NB, 1.0f);
}
